// TwoSideGraphModel_71717363909302
// MI455X (gfx1250) — compile-verified, matrix operand order checked by reading, not run on hardware
//
#include <hip/hip_runtime.h>
#include <hip/hip_bf16.h>

typedef __attribute__((ext_vector_type(16))) _Float16 v16h;
typedef __attribute__((ext_vector_type(8)))  float    v8f;

#define D 128
#define NEG_SLOPE 0.2f
#define EPS 1e-12f

union Frag { v16h v; unsigned u[8]; };

// ---------------------------------------------------------------------------
// Convert 6 weight matrices (128x128 f32, row-major [k][n]) to f16 transposed
// [n][k] so B-fragment K-pairs are contiguous 32-bit LDS/global loads.
// ---------------------------------------------------------------------------
__global__ void prep_weights(const float* __restrict__ w0, const float* __restrict__ w1,
                             const float* __restrict__ w2, const float* __restrict__ w3,
                             const float* __restrict__ w4, const float* __restrict__ w5,
                             _Float16* __restrict__ out) {
    int t = blockIdx.x * 256 + threadIdx.x;
    if (t >= 6 * D * D) return;
    int w = t / (D * D);
    int rem = t - w * (D * D);
    int n = rem >> 7, k = rem & 127;
    const float* W = (w == 0) ? w0 : (w == 1) ? w1 : (w == 2) ? w2
                   : (w == 3) ? w3 : (w == 4) ? w4 : w5;
    out[t] = (_Float16)W[k * D + n];
}

// ---------------------------------------------------------------------------
// Edge scatter: one wave per edge. Lane i moves float4 chunk i of the 128-f32
// source row; atomic f32 adds into agg[dst], degree count by lane 0.
// ---------------------------------------------------------------------------
__global__ void edge_scatter(const float* __restrict__ h, const int* __restrict__ src,
                             const int* __restrict__ dst, float* __restrict__ agg,
                             float* __restrict__ deg, int E) {
    unsigned t = blockIdx.x * 256u + threadIdx.x;
    int e = (int)(t >> 5);
    if (e >= E) return;
    int lane = (int)(t & 31u);
    int s = src[e], d = dst[e];
    const float4* hr = (const float4*)(h + (size_t)s * D);
    float4 v = hr[lane];
    float* ar = agg + (size_t)d * D + lane * 4;
    unsafeAtomicAdd(ar + 0, v.x);
    unsafeAtomicAdd(ar + 1, v.y);
    unsafeAtomicAdd(ar + 2, v.z);
    unsafeAtomicAdd(ar + 3, v.w);
    if (lane == 0) unsafeAtomicAdd(deg + d, 1.0f);
}

// ---------------------------------------------------------------------------
// Fused layer: out = l2norm(leakyrelu(h@Ws + (agg/deg)@Wn + b))
// Block = 256 threads (8 waves) computes a 16-row x 128-col tile.
// Wave w owns output columns [16w, 16w+16). K=128 swept in 4 chunks of 32.
// Weights are f16 transposed [n][k]. NEIGH=false -> plain dense layer.
// ---------------------------------------------------------------------------
template <bool NEIGH>
__global__ void fused_layer(const float* __restrict__ h, const float* __restrict__ agg,
                            const float* __restrict__ deg,
                            const _Float16* __restrict__ wself_t,
                            const _Float16* __restrict__ wneigh_t,
                            const float* __restrict__ bias,
                            float* __restrict__ out) {
    __shared__ _Float16 ldsA[16 * D];   // h tile (f16)
    __shared__ _Float16 ldsN[16 * D];   // neighbor tile (f16)
    __shared__ float    ldsO[16 * D];   // pre-norm output tile
    __shared__ float    ldsP[16 * 16];  // per-row partial sums of squares
    __shared__ float    ldsR[16];       // per-row 1/norm

    const int m0 = blockIdx.x * 16;
    const int tid = threadIdx.x;

    // Phase 1: stage tiles into LDS as f16
    for (int i = tid; i < 16 * D; i += 256) {
        int r = i >> 7, c = i & 127;
        int row = m0 + r;
        ldsA[i] = (_Float16)h[(size_t)row * D + c];
        if constexpr (NEIGH) {
            float dv = fmaxf(deg[row], 1.0f);
            ldsN[i] = (_Float16)(agg[(size_t)row * D + c] / dv);
        }
    }
    __syncthreads();

    // Phase 2: WMMA accumulation
    const int wave = tid >> 5;
    const int lane = tid & 31;
    const int grp  = lane >> 4;   // lane group (0: lanes 0-15, 1: lanes 16-31)
    const int ln   = lane & 15;
    const int n0   = wave * 16;

    v8f acc = {};
#pragma unroll
    for (int kc = 0; kc < 4; ++kc) {
        const int k0 = kc * 32;
        Frag ah, an, bs, bn;
#pragma unroll
        for (int v = 0; v < 8; ++v) {
            // ISA 7.12.2: 16-bit A 16x32 layout -> K-pair base per VGPR
            int bk = k0 + ((v < 4) ? 2 * v : 8 + 2 * v) + 8 * grp;
            ah.u[v] = *(const unsigned*)&ldsA[ln * D + bk];
            bs.u[v] = *(const unsigned*)&wself_t[(size_t)(n0 + ln) * D + bk];
            if constexpr (NEIGH) {
                an.u[v] = *(const unsigned*)&ldsN[ln * D + bk];
                bn.u[v] = *(const unsigned*)&wneigh_t[(size_t)(n0 + ln) * D + bk];
            }
        }
        acc = __builtin_amdgcn_wmma_f32_16x16x32_f16(false, ah.v, false, bs.v,
                                                     (short)0, acc, false, false);
        if constexpr (NEIGH) {
            acc = __builtin_amdgcn_wmma_f32_16x16x32_f16(false, an.v, false, bn.v,
                                                         (short)0, acc, false, false);
        }
    }

    // Phase 3: bias + leaky relu, spill tile to LDS (C/D layout: VGPR r -> row r or r+8)
#pragma unroll
    for (int r = 0; r < 8; ++r) {
        int row = r + grp * 8;
        int col = n0 + ln;
        float val = acc[r] + bias[col];
        val = (val > 0.0f) ? val : NEG_SLOPE * val;
        ldsO[row * D + col] = val;
    }
    __syncthreads();

    // Phase 4: row-wise L2 normalization, then store
    const int row = tid >> 4;       // 0..15
    const int seg = tid & 15;       // 0..15, 8 elems each
    const int c0  = seg * 8;
    float s = 0.0f;
#pragma unroll
    for (int i = 0; i < 8; ++i) {
        float x = ldsO[row * D + c0 + i];
        s += x * x;
    }
    ldsP[row * 16 + seg] = s;
    __syncthreads();
    if (tid < 16) {
        float ss = 0.0f;
#pragma unroll
        for (int j = 0; j < 16; ++j) ss += ldsP[tid * 16 + j];
        ldsR[tid] = 1.0f / fmaxf(sqrtf(ss), EPS);
    }
    __syncthreads();
    const float rn = ldsR[row];
    float* orow = out + (size_t)(m0 + row) * D + c0;
#pragma unroll
    for (int i = 0; i < 8; ++i) orow[i] = ldsO[row * D + c0 + i] * rn;
}

// ---------------------------------------------------------------------------
// Final gather: out[b, 0:128]=f0[idx], [128:256]=f1[idx], [256:384]=f2[idx]
// ---------------------------------------------------------------------------
__global__ void gather_out(const int* __restrict__ idx, const float* __restrict__ f0,
                           const float* __restrict__ f1, const float* __restrict__ f2,
                           float* __restrict__ out, int B) {
    int t = blockIdx.x * 256 + threadIdx.x;
    if (t >= B * 384) return;
    int row = t / 384;
    int c = t - row * 384;
    int node = idx[row];
    float v;
    if (c < 128)      v = f0[(size_t)node * D + c];
    else if (c < 256) v = f1[(size_t)node * D + (c - 128)];
    else              v = f2[(size_t)node * D + (c - 256)];
    out[t] = v;
}

extern "C" void kernel_launch(void* const* d_in, const int* in_sizes, int n_in,
                              void* d_out, int out_size, void* d_ws, size_t ws_size,
                              hipStream_t stream) {
    const int*   src       = (const int*)d_in[0];
    const int*   dst       = (const int*)d_in[1];
    const int*   users     = (const int*)d_in[2];
    const int*   pos_items = (const int*)d_in[3];
    const int*   neg_items = (const int*)d_in[4];
    const float* user_feat = (const float*)d_in[5];
    const float* item_feat = (const float*)d_in[6];
    const float* Wu_self0  = (const float*)d_in[7];
    const float* Wu_neigh0 = (const float*)d_in[8];
    const float* bu0       = (const float*)d_in[9];
    const float* Wi0       = (const float*)d_in[10];
    const float* bi0       = (const float*)d_in[11];
    const float* Wu_self1  = (const float*)d_in[12];
    const float* Wu_neigh1 = (const float*)d_in[13];
    const float* bu1       = (const float*)d_in[14];
    const float* Wi1       = (const float*)d_in[15];
    const float* bi1       = (const float*)d_in[16];

    const int E      = in_sizes[0];
    const int B      = in_sizes[2];
    const int N_USER = in_sizes[5] / D;
    const int N_ITEM = in_sizes[6] / D;

    // Workspace layout (agg aliased as h_i1 after the user tower finishes)
    char* ws = (char*)d_ws;
    const size_t featU = (size_t)N_USER * D * sizeof(float);
    const size_t featI = (size_t)N_ITEM * D * sizeof(float);
    size_t off = 0;
    float* agg  = (float*)(ws + off);               off += featI;  // reused as h_i1
    float* deg  = (float*)(ws + off);               off += ((size_t)N_USER * 4 + 255) & ~255ull;
    float* h_u1 = (float*)(ws + off);               off += featU;
    float* h_u2 = (float*)(ws + off);               off += featU;
    float* h_i2 = (float*)(ws + off);               off += featI;
    _Float16* wf16 = (_Float16*)(ws + off);         off += 6 * D * D * sizeof(_Float16);
    float* h_i1 = agg;

    _Float16* wtS0 = wf16 + 0 * D * D;  // Wu_self0^T
    _Float16* wtN0 = wf16 + 1 * D * D;  // Wu_neigh0^T
    _Float16* wtS1 = wf16 + 2 * D * D;  // Wu_self1^T
    _Float16* wtN1 = wf16 + 3 * D * D;  // Wu_neigh1^T
    _Float16* wtI0 = wf16 + 4 * D * D;  // Wi0^T
    _Float16* wtI1 = wf16 + 5 * D * D;  // Wi1^T

    // Weight conversion/transpose (f32 -> f16 [n][k])
    prep_weights<<<(6 * D * D + 255) / 256, 256, 0, stream>>>(
        Wu_self0, Wu_neigh0, Wu_self1, Wu_neigh1, Wi0, Wi1, wf16);

    const int scatterBlocks = (E * 32 + 255) / 256;
    const int tilesU = N_USER / 16;   // 100000/16 = 6250, exact
    const int tilesI = N_ITEM / 16;

    // ---- User tower, layer 0 ----
    hipMemsetAsync(agg, 0, featU + (size_t)N_USER * 4, stream);  // agg + deg contiguous
    edge_scatter<<<scatterBlocks, 256, 0, stream>>>(user_feat, src, dst, agg, deg, E);
    fused_layer<true><<<tilesU, 256, 0, stream>>>(user_feat, agg, deg, wtS0, wtN0, bu0, h_u1);

    // ---- User tower, layer 1 ----
    hipMemsetAsync(agg, 0, featU + (size_t)N_USER * 4, stream);
    edge_scatter<<<scatterBlocks, 256, 0, stream>>>(h_u1, src, dst, agg, deg, E);
    fused_layer<true><<<tilesU, 256, 0, stream>>>(h_u1, agg, deg, wtS1, wtN1, bu1, h_u2);

    // ---- Item tower (agg no longer needed -> reused as h_i1) ----
    fused_layer<false><<<tilesI, 256, 0, stream>>>(item_feat, nullptr, nullptr, wtI0, wtI0, bi0, h_i1);
    fused_layer<false><<<tilesI, 256, 0, stream>>>(h_i1, nullptr, nullptr, wtI1, wtI1, bi1, h_i2);

    // ---- Gathers: [users | pos | neg], each B x 384 ----
    float* out = (float*)d_out;
    const int gBlocks = (B * 384 + 255) / 256;
    gather_out<<<gBlocks, 256, 0, stream>>>(users,     user_feat, h_u1, h_u2, out,               B);
    gather_out<<<gBlocks, 256, 0, stream>>>(pos_items, item_feat, h_i1, h_i2, out + (size_t)B * 384,     B);
    gather_out<<<gBlocks, 256, 0, stream>>>(neg_items, item_feat, h_i1, h_i2, out + (size_t)2 * B * 384, B);
}
